// GRUDecoder_2319282340016
// MI455X (gfx1250) — matrix operand order repaired and checked by reading, not compiled, not run on hardware
//
#include <hip/hip_runtime.h>
#include <hip/hip_bf16.h>

typedef _Float16 half_t;
typedef __attribute__((ext_vector_type(16))) _Float16 v16h;
typedef __attribute__((ext_vector_type(8)))  float    v8f;

#define Hdim    128
#define Idim    128
#define Tsteps  30
#define Mmodes  6
#define Nagents 8192
#define Brows   (Mmodes * Nagents)

// ---------------------------------------------------------------------------
// WMMA helpers: D = A(16x32 f16) x B(32x16 f16) + C(16x16 f32), wave32.
// ---------------------------------------------------------------------------
__device__ __forceinline__ v8f wmma_step(v8f c, v16h a, v16h b) {
  return __builtin_amdgcn_wmma_f32_16x16x32_f16(
      /*neg_a=*/false, a, /*neg_b=*/false, b,
      /*c_mod=*/(short)0, c, /*reuse_a=*/false, /*reuse_b=*/false);
}

// A fragment (16x32, f16) from row-major LDS tile [rows][stride].
// ISA layout: lane row = lane&15; lane-half selects K {0..7,16..23} vs {8..15,24..31}.
__device__ __forceinline__ v16h load_a_frag(const half_t* base, int stride, int row0,
                                            int kbase, int l15, int hi) {
  const half_t* rp = base + (row0 + l15) * stride + kbase + hi * 8;
  v16h a;
#pragma unroll
  for (int v = 0; v < 8; ++v) {
    int K = ((v >> 2) << 4) + ((v & 3) << 1);
    a[2 * v]     = rp[K];
    a[2 * v + 1] = rp[K + 1];
  }
  return a;
}

// B fragment (32x16, f16) from weights stored row-major [out][in] (PyTorch layout),
// i.e. B[k][n] = W[n][k]. Lane col n, lane-half selects K {0..7,16..23} vs {8..15,24..31}.
__device__ __forceinline__ v16h load_b_frag(const half_t* w, int in_dim, int n,
                                            int kb, int hi) {
  const half_t* rp = w + n * in_dim + kb + hi * 8;
  v16h b;
#pragma unroll
  for (int v = 0; v < 16; ++v) b[v] = rp[v + ((v >> 3) << 3)];
  return b;
}

// [16 x in] @ W.T -> 8 col tiles, + bias, LayerNorm(128) + ReLU, all in registers.
// Row-wise LN reduction uses 16-lane xor shuffles (C-layout rows live per lane-half).
template <int NKT>
__device__ __forceinline__ void mm_ln_relu(v8f acc[8], const v16h* ha, const half_t* w,
                                           int in_dim, const float* bias,
                                           const float* gamma, const float* beta,
                                           int l15, int hi) {
#pragma unroll
  for (int ct = 0; ct < 8; ++ct) {
    float bv = bias[ct * 16 + l15];
    v8f a;
#pragma unroll
    for (int g = 0; g < 8; ++g) a[g] = bv;
#pragma unroll
    for (int kt = 0; kt < NKT; ++kt)
      a = wmma_step(a, ha[kt], load_b_frag(w, in_dim, ct * 16 + l15, kt * 32, hi));
    acc[ct] = a;
  }
#pragma unroll
  for (int g = 0; g < 8; ++g) {
    float s = 0.f, q = 0.f;
#pragma unroll
    for (int ct = 0; ct < 8; ++ct) {
      float v = acc[ct][g];
      s += v; q += v * v;
    }
#pragma unroll
    for (int msk = 1; msk < 16; msk <<= 1) {
      s += __shfl_xor(s, msk, 32);
      q += __shfl_xor(q, msk, 32);
    }
    float mean = s * 0.0078125f;
    float var  = q * 0.0078125f - mean * mean;
    float rs   = rsqrtf(var + 1e-5f);
#pragma unroll
    for (int ct = 0; ct < 8; ++ct) {
      float v = (acc[ct][g] - mean) * rs * gamma[ct * 16 + l15] + beta[ct * 16 + l15];
      acc[ct][g] = v > 0.f ? v : 0.f;
    }
  }
}

// ---------------------------------------------------------------------------
// Kernel 1: pi head.  pi_in[B,256] -> L(256,128)+LN+ReLU -> L(128,128)+LN+ReLU
//                     -> L(128,1) -> pi[N,M]
// ---------------------------------------------------------------------------
__global__ void __launch_bounds__(256, 1) pi_head_kernel(
    const float* __restrict__ local_embed, const float* __restrict__ global_embed,
    const float* __restrict__ pi1_w, const float* __restrict__ pi1_b,
    const float* __restrict__ ln1_g, const float* __restrict__ ln1_b,
    const float* __restrict__ pi2_w, const float* __restrict__ pi2_b,
    const float* __restrict__ ln2_g, const float* __restrict__ ln2_b,
    const float* __restrict__ pi3_w, const float* __restrict__ pi3_b,
    float* __restrict__ pi_out) {
  extern __shared__ char smem[];
  half_t* w1 = (half_t*)smem;         // 128*256 f16
  half_t* w2 = w1 + 128 * 256;        // 128*128 f16
  half_t* xs = w2 + 128 * 128;        // 128*256 f16 (pi_in, reused for h1 stage)
  float* fs   = (float*)(xs + 128 * 256);
  float* s_b1 = fs;                   // 128
  float* s_g1 = s_b1 + 128; float* s_t1 = s_g1 + 128;
  float* s_b2 = s_t1 + 128; float* s_g2 = s_b2 + 128; float* s_t2 = s_g2 + 128;
  float* s_w3 = s_t2 + 128;           // 128

  const int tid = threadIdx.x;
  const int lane = tid & 31, wave = tid >> 5;
  const int l15 = lane & 15, hi = lane >> 4;
  const int row0 = wave * 16;
  const long blockRow = (long)blockIdx.x * 128;

  for (int i = tid; i < 128 * 256; i += 256) w1[i] = (half_t)pi1_w[i];
  for (int i = tid; i < 128 * 128; i += 256) w2[i] = (half_t)pi2_w[i];
  for (int i = tid; i < 128 * 256; i += 256) {
    int r = i >> 8, c = i & 255;
    long b = blockRow + r;
    float v = (c < 128) ? local_embed[(b & (Nagents - 1)) * 128 + c]
                        : global_embed[b * 128 + (c - 128)];
    xs[i] = (half_t)v;
  }
  for (int i = tid; i < 128; i += 256) {
    s_b1[i] = pi1_b[i]; s_g1[i] = ln1_g[i]; s_t1[i] = ln1_b[i];
    s_b2[i] = pi2_b[i]; s_g2[i] = ln2_g[i]; s_t2[i] = ln2_b[i];
    s_w3[i] = pi3_w[i];
  }
  __syncthreads();

  v16h a8[8];
#pragma unroll
  for (int kt = 0; kt < 8; ++kt) a8[kt] = load_a_frag(xs, 256, row0, kt * 32, l15, hi);
  v8f acc[8];
  mm_ln_relu<8>(acc, a8, w1, 256, s_b1, s_g1, s_t1, l15, hi);

  // stage h1 (f16) into our private slab (cols 0..127, stride 256)
#pragma unroll
  for (int ct = 0; ct < 8; ++ct)
#pragma unroll
    for (int g = 0; g < 8; ++g)
      xs[(row0 + g + 8 * hi) * 256 + ct * 16 + l15] = (half_t)acc[ct][g];

  v16h hb[4];
#pragma unroll
  for (int kt = 0; kt < 4; ++kt) hb[kt] = load_a_frag(xs, 256, row0, kt * 32, l15, hi);
  mm_ln_relu<4>(acc, hb, w2, 128, s_b2, s_g2, s_t2, l15, hi);

  const float b3 = pi3_b[0];
#pragma unroll
  for (int g = 0; g < 8; ++g) {
    float p = 0.f;
#pragma unroll
    for (int ct = 0; ct < 8; ++ct) p += acc[ct][g] * s_w3[ct * 16 + l15];
#pragma unroll
    for (int msk = 1; msk < 16; msk <<= 1) p += __shfl_xor(p, msk, 32);
    if (l15 == 0) {
      long b = blockRow + row0 + g + 8 * hi;
      int m = (int)(b >> 13);             // b / N   (N = 8192)
      int n = (int)(b & (Nagents - 1));   // b % N
      pi_out[(long)n * Mmodes + m] = p + b3;
    }
  }
}

// ---------------------------------------------------------------------------
// Kernel 2: fused  gx precompute + GRU(T=30) + loc/sc heads -> traj[M,N,T,4].
// Hidden state lives in LDS (f16) for the whole recurrence; out[B,T,H] is
// never materialized in HBM.
// ---------------------------------------------------------------------------
__global__ void __launch_bounds__(256, 1) gru_traj_kernel(
    const float* __restrict__ local_embed, const float* __restrict__ global_embed,
    const float* __restrict__ W_ih, const float* __restrict__ W_hh,
    const float* __restrict__ b_ih, const float* __restrict__ b_hh,
    const float* __restrict__ loc1_w, const float* __restrict__ loc1_b,
    const float* __restrict__ loc_ln_g, const float* __restrict__ loc_ln_b,
    const float* __restrict__ loc2_w, const float* __restrict__ loc2_b,
    const float* __restrict__ sc1_w, const float* __restrict__ sc1_b,
    const float* __restrict__ sc_ln_g, const float* __restrict__ sc_ln_b,
    const float* __restrict__ sc2_w, const float* __restrict__ sc2_b,
    float* __restrict__ traj_out) {
  extern __shared__ char smem[];
  half_t* wbuf = (half_t*)smem;       // 384*128 f16 : W_ih, then W_hh
  half_t* gx   = wbuf + 384 * 128;    // 128*384 f16 : x@W_ih.T + b_ih
  half_t* hbuf = gx + 128 * 384;      // 128*128 f16 : x tile, then hidden state
  half_t* wloc = hbuf + 128 * 128;    // 128*128 f16
  half_t* wsc  = wloc + 128 * 128;    // 128*128 f16
  float* fs      = (float*)(wsc + 128 * 128);
  float* tstage  = fs;                // 8 waves * 16 rows * 4 = 512
  float* s_bih   = tstage + 512;      // 384
  float* s_bhh   = s_bih + 384;       // 384
  float* s_l1b   = s_bhh + 384;       // 128
  float* s_lg    = s_l1b + 128; float* s_lb = s_lg + 128;
  float* s_l2w   = s_lb + 128;        // 256
  float* s_s1b   = s_l2w + 256; float* s_sg = s_s1b + 128; float* s_sb = s_sg + 128;
  float* s_s2w   = s_sb + 128;        // 256

  const int tid = threadIdx.x;
  const int lane = tid & 31, wave = tid >> 5;
  const int l15 = lane & 15, hi = lane >> 4;
  const int row0 = wave * 16;
  const long blockRow = (long)blockIdx.x * 128;

  // ---- prologue A: stage W_ih + x tile + small params ----
  for (int i = tid; i < 384 * 128; i += 256) wbuf[i] = (half_t)W_ih[i];
  for (int i = tid; i < 128 * 128; i += 256)
    hbuf[i] = (half_t)global_embed[blockRow * 128 + i];
  for (int i = tid; i < 384; i += 256) { s_bih[i] = b_ih[i]; s_bhh[i] = b_hh[i]; }
  for (int i = tid; i < 128; i += 256) {
    s_l1b[i] = loc1_b[i]; s_lg[i] = loc_ln_g[i]; s_lb[i] = loc_ln_b[i];
    s_s1b[i] = sc1_b[i];  s_sg[i] = sc_ln_g[i];  s_sb[i] = sc_ln_b[i];
  }
  for (int i = tid; i < 256; i += 256) { s_l2w[i] = loc2_w[i]; s_s2w[i] = sc2_w[i]; }
  __syncthreads();

  // ---- gx = x @ W_ih.T + b_ih  (input gates, same every timestep) ----
  {
    v16h xa[4];
#pragma unroll
    for (int kt = 0; kt < 4; ++kt) xa[kt] = load_a_frag(hbuf, 128, row0, kt * 32, l15, hi);
#pragma unroll 1
    for (int ct = 0; ct < 24; ++ct) {
      float bi = s_bih[ct * 16 + l15];
      v8f a;
#pragma unroll
      for (int g = 0; g < 8; ++g) a[g] = bi;
#pragma unroll
      for (int kt = 0; kt < 4; ++kt)
        a = wmma_step(a, xa[kt], load_b_frag(wbuf, 128, ct * 16 + l15, kt * 32, hi));
#pragma unroll
      for (int g = 0; g < 8; ++g)
        gx[(row0 + g + 8 * hi) * 384 + ct * 16 + l15] = (half_t)a[g];
    }
  }
  __syncthreads();

  // ---- prologue B: overwrite with W_hh, h0 = local_embed, head weights ----
  for (int i = tid; i < 384 * 128; i += 256) wbuf[i] = (half_t)W_hh[i];
  for (int i = tid; i < 128 * 128; i += 256) {
    int r = i >> 7, c = i & 127;
    long b = blockRow + r;
    hbuf[i] = (half_t)local_embed[(b & (Nagents - 1)) * 128 + c];
  }
  for (int i = tid; i < 128 * 128; i += 256) {
    wloc[i] = (half_t)loc1_w[i];
    wsc[i]  = (half_t)sc1_w[i];
  }
  __syncthreads();

  const float l2b0 = loc2_b[0], l2b1 = loc2_b[1];
  const float s2b0 = sc2_b[0],  s2b1 = sc2_b[1];

  // ---- recurrence + fused heads ----
  for (int t = 0; t < Tsteps; ++t) {
    v16h ha[4];
#pragma unroll
    for (int kt = 0; kt < 4; ++kt) ha[kt] = load_a_frag(hbuf, 128, row0, kt * 32, l15, hi);

#pragma unroll 1
    for (int j = 0; j < 8; ++j) {  // 16-col tile of the hidden state
      float br = s_bhh[j * 16 + l15];
      float bz = s_bhh[128 + j * 16 + l15];
      float bn = s_bhh[256 + j * 16 + l15];
      v8f ar, az, an;
#pragma unroll
      for (int g = 0; g < 8; ++g) { ar[g] = br; az[g] = bz; an[g] = bn; }
#pragma unroll
      for (int kt = 0; kt < 4; ++kt) {
        ar = wmma_step(ar, ha[kt], load_b_frag(wbuf, 128,       j * 16 + l15, kt * 32, hi));
        az = wmma_step(az, ha[kt], load_b_frag(wbuf, 128, 128 + j * 16 + l15, kt * 32, hi));
        an = wmma_step(an, ha[kt], load_b_frag(wbuf, 128, 256 + j * 16 + l15, kt * 32, hi));
      }
#pragma unroll
      for (int g = 0; g < 8; ++g) {
        int m = row0 + g + 8 * hi;
        int c = j * 16 + l15;
        float xr   = (float)gx[m * 384 + c];
        float xz   = (float)gx[m * 384 + 128 + c];
        float xn   = (float)gx[m * 384 + 256 + c];
        float hold = (float)hbuf[m * 128 + c];
        float r = 1.f / (1.f + __expf(-(xr + ar[g])));
        float z = 1.f / (1.f + __expf(-(xz + az[g])));
        float n = tanhf(xn + r * an[g]);
        hbuf[m * 128 + c] = (half_t)((1.f - z) * n + z * hold);
      }
    }

    // heads on h_new (re-fragment from LDS)
#pragma unroll
    for (int kt = 0; kt < 4; ++kt) ha[kt] = load_a_frag(hbuf, 128, row0, kt * 32, l15, hi);

    v8f acc[8];
    mm_ln_relu<4>(acc, ha, wloc, 128, s_l1b, s_lg, s_lb, l15, hi);
#pragma unroll
    for (int g = 0; g < 8; ++g) {
      float px = 0.f, py = 0.f;
#pragma unroll
      for (int ct = 0; ct < 8; ++ct) {
        float v = acc[ct][g];
        px += v * s_l2w[ct * 16 + l15];
        py += v * s_l2w[128 + ct * 16 + l15];
      }
#pragma unroll
      for (int msk = 1; msk < 16; msk <<= 1) {
        px += __shfl_xor(px, msk, 32);
        py += __shfl_xor(py, msk, 32);
      }
      int rloc = wave * 64 + (g + 8 * hi) * 4;
      if (l15 == 0) tstage[rloc + 0] = px + l2b0;
      if (l15 == 1) tstage[rloc + 1] = py + l2b1;
    }

    mm_ln_relu<4>(acc, ha, wsc, 128, s_s1b, s_sg, s_sb, l15, hi);
#pragma unroll
    for (int g = 0; g < 8; ++g) {
      float px = 0.f, py = 0.f;
#pragma unroll
      for (int ct = 0; ct < 8; ++ct) {
        float v = acc[ct][g];
        px += v * s_s2w[ct * 16 + l15];
        py += v * s_s2w[128 + ct * 16 + l15];
      }
#pragma unroll
      for (int msk = 1; msk < 16; msk <<= 1) {
        px += __shfl_xor(px, msk, 32);
        py += __shfl_xor(py, msk, 32);
      }
      px += s2b0; py += s2b1;
      // elu(x) + 1 + MIN_SCALE
      px = (px > 0.f) ? (px + 1.001f) : (__expf(px) + 0.001f);
      py = (py > 0.f) ? (py + 1.001f) : (__expf(py) + 0.001f);
      int rloc = wave * 64 + (g + 8 * hi) * 4;
      if (l15 == 0) tstage[rloc + 2] = px;
      if (l15 == 1) tstage[rloc + 3] = py;
    }

    // traj[b, t, 0:4] as one float4 per row
    if (lane < 16) {
      long b = blockRow + row0 + lane;
      const float4 vv = *(const float4*)&tstage[wave * 64 + lane * 4];
      *(float4*)&traj_out[(b * Tsteps + t) * 4] = vv;
    }
  }
}

// ---------------------------------------------------------------------------
extern "C" void kernel_launch(void* const* d_in, const int* in_sizes, int n_in,
                              void* d_out, int out_size, void* d_ws, size_t ws_size,
                              hipStream_t stream) {
  (void)in_sizes; (void)n_in; (void)out_size; (void)d_ws; (void)ws_size;
  const float* local_embed  = (const float*)d_in[0];
  const float* global_embed = (const float*)d_in[1];
  const float* W_ih   = (const float*)d_in[2];
  const float* W_hh   = (const float*)d_in[3];
  const float* b_ih   = (const float*)d_in[4];
  const float* b_hh   = (const float*)d_in[5];
  const float* loc1_w = (const float*)d_in[6];
  const float* loc1_b = (const float*)d_in[7];
  const float* loc_ln_g = (const float*)d_in[8];
  const float* loc_ln_b = (const float*)d_in[9];
  const float* loc2_w = (const float*)d_in[10];
  const float* loc2_b = (const float*)d_in[11];
  const float* sc1_w  = (const float*)d_in[12];
  const float* sc1_b  = (const float*)d_in[13];
  const float* sc_ln_g = (const float*)d_in[14];
  const float* sc_ln_b = (const float*)d_in[15];
  const float* sc2_w  = (const float*)d_in[16];
  const float* sc2_b  = (const float*)d_in[17];
  const float* pi1_w  = (const float*)d_in[18];
  const float* pi1_b  = (const float*)d_in[19];
  const float* pi_ln1_g = (const float*)d_in[20];
  const float* pi_ln1_b = (const float*)d_in[21];
  const float* pi2_w  = (const float*)d_in[22];
  const float* pi2_b  = (const float*)d_in[23];
  const float* pi_ln2_g = (const float*)d_in[24];
  const float* pi_ln2_b = (const float*)d_in[25];
  const float* pi3_w  = (const float*)d_in[26];
  const float* pi3_b  = (const float*)d_in[27];

  float* traj = (float*)d_out;
  float* pi   = traj + (long)Mmodes * Nagents * Tsteps * 4;

  constexpr int LDS_PI  = (128 * 256 + 128 * 128 + 128 * 256) * 2 + 896 * 4;   // 167,424 B
  constexpr int LDS_GRU = (384 * 128 + 128 * 384 + 3 * 128 * 128) * 2 + 2560 * 4; // 305,152 B

  (void)hipFuncSetAttribute((const void*)pi_head_kernel,
                            hipFuncAttributeMaxDynamicSharedMemorySize, LDS_PI);
  (void)hipFuncSetAttribute((const void*)gru_traj_kernel,
                            hipFuncAttributeMaxDynamicSharedMemorySize, LDS_GRU);

  const int nblocks = Brows / 128;  // 384
  pi_head_kernel<<<nblocks, 256, LDS_PI, stream>>>(
      local_embed, global_embed, pi1_w, pi1_b, pi_ln1_g, pi_ln1_b,
      pi2_w, pi2_b, pi_ln2_g, pi_ln2_b, pi3_w, pi3_b, pi);
  gru_traj_kernel<<<nblocks, 256, LDS_GRU, stream>>>(
      local_embed, global_embed, W_ih, W_hh, b_ih, b_hh,
      loc1_w, loc1_b, loc_ln_g, loc_ln_b, loc2_w, loc2_b,
      sc1_w, sc1_b, sc_ln_g, sc_ln_b, sc2_w, sc2_b, traj);
}
